// ViT_TcssAMR_37031208026447
// MI455X (gfx1250) — matrix operand order repaired and checked by reading, not compiled, not run on hardware
//
#include <hip/hip_runtime.h>
#include <cstdint>
#include <cstddef>

// ---------------- problem constants ----------------
constexpr int NB = 4, NL = 1024, NDIM = 256, NHEADS = 8, NDH = 64, NDEPTH = 2,
              NMLP = 512, NINNER = 512;
constexpr int MTOK = NB * NL;  // 4096 tokens

// ---------------- types ----------------
typedef __bf16 bf16_t;
typedef __attribute__((ext_vector_type(16))) __bf16       v16bf;
typedef __attribute__((ext_vector_type(8)))  float        v8f;
typedef __attribute__((ext_vector_type(4)))  unsigned int u32x4;

union FragBF { u32x4 q[2]; v16bf v; };  // 32 bytes, trivial members

__device__ __forceinline__ bf16_t to_bf16(float f) {
  unsigned u = __float_as_uint(f);
  u += 0x7FFFu + ((u >> 16) & 1u);                  // round-to-nearest-even
  unsigned short h = (unsigned short)(u >> 16);
  return __builtin_bit_cast(bf16_t, h);
}
__device__ __forceinline__ bf16_t bf16_zero() {
  return __builtin_bit_cast(bf16_t, (unsigned short)0);
}

__device__ __forceinline__ v8f wmma_bf16(const FragBF& a, const FragBF& b, v8f c) {
  return __builtin_amdgcn_wmma_f32_16x16x32_bf16(false, a.v, false, b.v,
                                                 (short)0, c, false, false);
}

// =====================================================================
// General batched GEMM:  C[z] = act(scale * A[z] * Bt[z]^T + bias + addmat[z]) + resid
// A: [M,K] row-major bf16 (lda), Bt: [N,K] row-major bf16 (ldb).
// Block = 4 waves; each wave owns a 16(M)x64(N) strip: 4 accumulators sharing
// one A fragment per K-step (V_WMMA_F32_16X16X32_BF16).  Requires M%64==0,
// N%64==0, K%32==0 (true for every call site here).
// Fragment layouts per CDNA5 ISA 7.12.2:
//   A: lane m=lane%16, K-base = (lane<16?0:8); elems 0..7 = K kb..kb+7, 8..15 = K kb+16..kb+23
//   B: lane n=lane%16, K-base = (lane<16?0:8); elems 0..7 = K kb..kb+7, 8..15 = K kb+16..kb+23
// =====================================================================
__global__ __launch_bounds__(128) void gemm_bf16_wmma(
    const bf16_t* __restrict__ A, int lda, long long zsA,
    const bf16_t* __restrict__ Bt, int ldb, long long zsB,
    float* __restrict__ Cf, bf16_t* __restrict__ Cb, int ldc, long long zsC,
    const float* __restrict__ bias, const float* __restrict__ resid,
    const float* __restrict__ addmat, long long zsAdd,
    float scale, int M, int N, int K, int act)
{
  const int wave = threadIdx.x >> 5;
  const int lane = threadIdx.x & 31;
  const int m0 = (blockIdx.y << 6) + (wave << 4);  // wave-private 16-row strip
  const int n0 = blockIdx.x << 6;                  // 64-col strip (4 tiles)
  if (m0 >= M || n0 >= N) return;
  const long long z = blockIdx.z;
  A  += z * zsA;
  Bt += z * zsB;

  const int half = lane >> 4;
  const bf16_t* ap = A  + (size_t)(m0 + (lane & 15)) * lda + (half << 3);
  const bf16_t* bp = Bt + (size_t)(n0 + (lane & 15)) * ldb + (half << 3);
  const size_t bstep = (size_t)16 * ldb;  // next 16-column tile of Bt

  const v8f vzero = {0.f, 0.f, 0.f, 0.f, 0.f, 0.f, 0.f, 0.f};
  v8f acc0 = vzero, acc1 = vzero, acc2 = vzero, acc3 = vzero;

  for (int k0 = 0; k0 < K; k0 += 32) {
    FragBF a, b0, b1, b2, b3;
    a.q[0]  = *reinterpret_cast<const u32x4*>(ap);
    a.q[1]  = *reinterpret_cast<const u32x4*>(ap + 16);
    b0.q[0] = *reinterpret_cast<const u32x4*>(bp);
    b0.q[1] = *reinterpret_cast<const u32x4*>(bp + 16);
    b1.q[0] = *reinterpret_cast<const u32x4*>(bp + bstep);
    b1.q[1] = *reinterpret_cast<const u32x4*>(bp + bstep + 16);
    b2.q[0] = *reinterpret_cast<const u32x4*>(bp + 2 * bstep);
    b2.q[1] = *reinterpret_cast<const u32x4*>(bp + 2 * bstep + 16);
    b3.q[0] = *reinterpret_cast<const u32x4*>(bp + 3 * bstep);
    b3.q[1] = *reinterpret_cast<const u32x4*>(bp + 3 * bstep + 16);
    __builtin_prefetch(ap + 64, 0, 1);              // global_prefetch_b8
    __builtin_prefetch(bp + 64, 0, 1);
    __builtin_prefetch(bp + bstep + 64, 0, 1);
    __builtin_prefetch(bp + 2 * bstep + 64, 0, 1);
    __builtin_prefetch(bp + 3 * bstep + 64, 0, 1);
    ap += 32;
    bp += 32;
    acc0 = wmma_bf16(a, b0, acc0);
    acc1 = wmma_bf16(a, b1, acc1);
    acc2 = wmma_bf16(a, b2, acc2);
    acc3 = wmma_bf16(a, b3, acc3);
  }

  // C layout: VGPR r -> row m0 + r + (lane<16?0:8), col = ntile*16 + lane%16
  const int rbase = m0 + (half << 3);
  const v8f accs[4] = {acc0, acc1, acc2, acc3};
#pragma unroll
  for (int tn = 0; tn < 4; ++tn) {
    const int col = n0 + (tn << 4) + (lane & 15);
    const float bc = bias ? bias[col] : 0.0f;
    const v8f acc = accs[tn];
#pragma unroll
    for (int r = 0; r < 8; ++r) {
      const int row = rbase + r;
      float v = acc[r] * scale + bc;
      if (addmat) v += addmat[z * zsAdd + (size_t)row * ldc + col];
      if (act == 1)      v = fmaxf(v, 0.0f);                                // ReLU
      else if (act == 2) v = 0.5f * v * (1.0f + erff(v * 0.70710678118f));  // exact GELU
      const size_t oidx = (size_t)(z * zsC) + (size_t)row * ldc + col;
      if (resid) v += resid[oidx];
      if (Cf) Cf[oidx] = v;
      if (Cb) Cb[oidx] = to_bf16(v);
    }
  }
}

// ---------------- weight prep ----------------
__global__ __launch_bounds__(256) void cvt_bf16_kernel(const float* __restrict__ in,
                                                       bf16_t* __restrict__ out, long long n) {
  long long i = (long long)blockIdx.x * 256 + threadIdx.x;
  if (i < n) out[i] = to_bf16(in[i]);
}
__global__ __launch_bounds__(256) void cvt_bf16_T_kernel(const float* __restrict__ in,
                                                         bf16_t* __restrict__ out, int R, int C) {
  long long i = (long long)blockIdx.x * 256 + threadIdx.x;
  if (i >= (long long)R * C) return;
  int r = (int)(i / C), c = (int)(i % C);
  out[(size_t)c * R + r] = to_bf16(in[i]);
}

// ---------------- conv patch embed ----------------
// img [B,1,2,L] -> y1 [B,2,L,256] f32  (1x3 conv, pad 1, relu)
__global__ __launch_bounds__(256) void conv1_kernel(const float* __restrict__ img,
                                                    const float* __restrict__ w,
                                                    const float* __restrict__ bias,
                                                    float* __restrict__ y1) {
  long long i = (long long)blockIdx.x * 256 + threadIdx.x;
  if (i >= (long long)NB * 2 * NL * 256) return;
  int c = i & 255, wp = (int)((i >> 8) & 1023), h = (int)((i >> 18) & 1), b = (int)(i >> 19);
  float s = bias[c];
#pragma unroll
  for (int kw = 0; kw < 3; ++kw) {
    int iw = wp + kw - 1;
    if (iw >= 0 && iw < NL) s += img[(size_t)(b * 2 + h) * NL + iw] * w[c * 3 + kw];
  }
  y1[i] = fmaxf(s, 0.0f);
}
// im2col for conv2 (2x3 kernel over 256ch): A2[b*L+w][(ci*2+kh)*3+kw]
__global__ __launch_bounds__(256) void im2col2_kernel(const float* __restrict__ y1,
                                                      bf16_t* __restrict__ A2) {
  long long i = (long long)blockIdx.x * 256 + threadIdx.x;
  if (i >= (long long)MTOK * 1536) return;
  int col = (int)(i % 1536); long long row = i / 1536;
  int wp = (int)(row & 1023), b = (int)(row >> 10);
  int kw = col % 3, kh = (col / 3) & 1, ci = col / 6;
  int iw = wp + kw - 1;
  float v = 0.0f;
  if (iw >= 0 && iw < NL) v = y1[((size_t)(b * 2 + kh) * NL + iw) * 256 + ci];
  A2[i] = to_bf16(v);
}
// im2col for conv3 (1x3 over 256ch): A3[b*L+w][ci*3+kw] from y2b [B*L,256] bf16
__global__ __launch_bounds__(256) void im2col3_kernel(const bf16_t* __restrict__ y2b,
                                                      bf16_t* __restrict__ A3) {
  long long i = (long long)blockIdx.x * 256 + threadIdx.x;
  if (i >= (long long)MTOK * 768) return;
  int col = (int)(i % 768); long long row = i / 768;
  int wp = (int)(row & 1023), b = (int)(row >> 10);
  int kw = col % 3, ci = col / 3;
  int iw = wp + kw - 1;
  A3[i] = (iw >= 0 && iw < NL) ? y2b[((size_t)b * NL + iw) * 256 + ci] : bf16_zero();
}

// ---------------- layernorm (row=token, 256 features) -> bf16 ----------------
__global__ __launch_bounds__(256) void layernorm_bf16_kernel(const float* __restrict__ x,
                                                             const float* __restrict__ g,
                                                             const float* __restrict__ bt,
                                                             bf16_t* __restrict__ out) {
  __shared__ float red[256];
  const int row = blockIdx.x, t = threadIdx.x;
  float v = x[(size_t)row * 256 + t];
  red[t] = v; __syncthreads();
  for (int s = 128; s > 0; s >>= 1) { if (t < s) red[t] += red[t + s]; __syncthreads(); }
  float mean = red[0] * (1.0f / 256.0f); __syncthreads();
  float d = v - mean;
  red[t] = d * d; __syncthreads();
  for (int s = 128; s > 0; s >>= 1) { if (t < s) red[t] += red[t + s]; __syncthreads(); }
  float var = red[0] * (1.0f / 256.0f);
  out[(size_t)row * 256 + t] = to_bf16(d * rsqrtf(var + 1e-5f) * g[t] + bt[t]);
}

// ---------------- V transpose (LDS-tiled, coalesced both sides) -------------
// vt[b][h][d][l] = qkvb[(b*L+l)*1536 + 1024 + h*64 + d]
// grid = (L/64, HEADS, B), block = 256.  64x64 bf16 tile, stride 66 vs banks.
__global__ __launch_bounds__(256) void build_vt_kernel(const bf16_t* __restrict__ qkvb,
                                                       bf16_t* __restrict__ vt) {
  __shared__ unsigned short tile[64 * 66];
  const int l0 = blockIdx.x << 6;
  const int h  = blockIdx.y;
  const int b  = blockIdx.z;
  const int t  = threadIdx.x;
#pragma unroll
  for (int i = 0; i < 16; ++i) {
    int idx = t + 256 * i;           // 0..4095
    int l = idx >> 6, d = idx & 63;  // consecutive t -> consecutive d (coalesced read)
    tile[l * 66 + d] = __builtin_bit_cast(unsigned short,
        qkvb[((size_t)(b * NL + l0 + l)) * 1536 + 1024 + h * 64 + d]);
  }
  __syncthreads();
#pragma unroll
  for (int i = 0; i < 16; ++i) {
    int idx = t + 256 * i;
    int d = idx >> 6, l = idx & 63;  // consecutive t -> consecutive l (coalesced write)
    vt[((size_t)(b * NHEADS + h) * NDH + d) * NL + l0 + l] =
        __builtin_bit_cast(bf16_t, tile[l * 66 + d]);
  }
}

// ---------------- softmax over rows of 1024 (f32 in, bf16 out) ----------------
__global__ __launch_bounds__(256) void softmax_rows_kernel(const float* __restrict__ S,
                                                           bf16_t* __restrict__ P) {
  __shared__ float red[256];
  const int row = blockIdx.x, t = threadIdx.x;
  const float* sr = S + (size_t)row * 1024;
  bf16_t* pr = P + (size_t)row * 1024;
  float v[4], lmax = -3.4e38f;
#pragma unroll
  for (int i = 0; i < 4; ++i) { v[i] = sr[t + 256 * i]; lmax = fmaxf(lmax, v[i]); }
  red[t] = lmax; __syncthreads();
  for (int s = 128; s > 0; s >>= 1) { if (t < s) red[t] = fmaxf(red[t], red[t + s]); __syncthreads(); }
  float m = red[0]; __syncthreads();
  float lsum = 0.0f;
#pragma unroll
  for (int i = 0; i < 4; ++i) { v[i] = __expf(v[i] - m); lsum += v[i]; }
  red[t] = lsum; __syncthreads();
  for (int s = 128; s > 0; s >>= 1) { if (t < s) red[t] += red[t + s]; __syncthreads(); }
  float inv = 1.0f / red[0];
#pragma unroll
  for (int i = 0; i < 4; ++i) pr[t + 256 * i] = to_bf16(v[i] * inv);
}

// ---------------- pooling ----------------
__global__ __launch_bounds__(256) void rowmean_kernel(const float* __restrict__ x,
                                                      float* __restrict__ w) {
  __shared__ float red[256];
  const int row = blockIdx.x, t = threadIdx.x;
  red[t] = x[(size_t)row * 256 + t]; __syncthreads();
  for (int s = 128; s > 0; s >>= 1) { if (t < s) red[t] += red[t + s]; __syncthreads(); }
  if (t == 0) w[row] = red[0] * (1.0f / 256.0f);
}
__global__ __launch_bounds__(128) void se_kernel(const float* __restrict__ wmean,
                                                 const float* __restrict__ w1,
                                                 const float* __restrict__ b1,
                                                 const float* __restrict__ w2,
                                                 const float* __restrict__ b2,
                                                 float* __restrict__ sgate) {
  __shared__ float wrow[1024];
  __shared__ float hid[128];
  const int b = blockIdx.x, t = threadIdx.x;
  for (int i = t; i < 1024; i += 128) wrow[i] = wmean[b * 1024 + i];
  __syncthreads();
  float a = b1[t];
  for (int l = 0; l < 1024; ++l) a += wrow[l] * w1[l * 128 + t];
  hid[t] = fmaxf(a, 0.0f);
  __syncthreads();
  for (int o = t; o < 1024; o += 128) {
    float a2 = b2[o];
    for (int j = 0; j < 128; ++j) a2 += hid[j] * w2[j * 1024 + o];
    sgate[b * 1024 + o] = 1.0f / (1.0f + __expf(-a2));
  }
}
__global__ __launch_bounds__(256) void pool_out_kernel(const float* __restrict__ x,
                                                       const float* __restrict__ s,
                                                       float* __restrict__ out) {
  __shared__ float sl[1024];
  const int b = blockIdx.x, d = threadIdx.x;
  for (int i = d; i < 1024; i += 256) sl[i] = s[b * 1024 + i];
  __syncthreads();
  float acc = 0.0f;
  for (int l = 0; l < 1024; ++l) acc += sl[l] * x[((size_t)b * 1024 + l) * 256 + d];
  out[b * 256 + d] = acc;
}

// ---------------- host-side launch helpers ----------------
static void launch_gemm(hipStream_t st,
                        const bf16_t* A, int lda, long long zsA,
                        const bf16_t* Bt, int ldb, long long zsB,
                        float* Cf, bf16_t* Cb, int ldc, long long zsC,
                        const float* bias, const float* resid,
                        const float* addmat, long long zsAdd,
                        float scale, int M, int N, int K, int Z, int act) {
  dim3 grid((N + 63) / 64, (M + 63) / 64, Z);
  gemm_bf16_wmma<<<grid, 128, 0, st>>>(A, lda, zsA, Bt, ldb, zsB, Cf, Cb, ldc, zsC,
                                       bias, resid, addmat, zsAdd, scale, M, N, K, act);
}

extern "C" void kernel_launch(void* const* d_in, const int* in_sizes, int n_in,
                              void* d_out, int out_size, void* d_ws, size_t ws_size,
                              hipStream_t stream) {
  (void)in_sizes; (void)n_in; (void)out_size; (void)ws_size;
  const float* img     = (const float*)d_in[0];
  const float* conv1_w = (const float*)d_in[1];
  const float* conv1_b = (const float*)d_in[2];
  const float* conv2_w = (const float*)d_in[3];
  const float* conv2_b = (const float*)d_in[4];
  const float* conv3_w = (const float*)d_in[5];
  const float* conv3_b = (const float*)d_in[6];
  const float* ln1_s   = (const float*)d_in[7];
  const float* ln1_b   = (const float*)d_in[8];
  const float* qkv_w   = (const float*)d_in[9];
  const float* pos     = (const float*)d_in[10];
  const float* out_w   = (const float*)d_in[11];
  const float* out_b   = (const float*)d_in[12];
  const float* ln2_s   = (const float*)d_in[13];
  const float* ln2_b   = (const float*)d_in[14];
  const float* ff_w1   = (const float*)d_in[15];
  const float* ff_b1   = (const float*)d_in[16];
  const float* ff_w2   = (const float*)d_in[17];
  const float* ff_b2   = (const float*)d_in[18];
  const float* se_w1   = (const float*)d_in[19];
  const float* se_b1   = (const float*)d_in[20];
  const float* se_w2   = (const float*)d_in[21];
  const float* se_b2   = (const float*)d_in[22];
  float* outp = (float*)d_out;

  // ---- workspace carve-up (all 256B aligned) ----
  char* ws = (char*)d_ws;
  size_t off = 0;
  auto alloc = [&](size_t bytes) -> char* {
    char* p = ws + off;
    off += (bytes + 255) & ~(size_t)255;
    return p;
  };
  float*  x      = (float*) alloc((size_t)MTOK * 256 * 4);          // residual stream f32
  bf16_t* hb     = (bf16_t*)alloc((size_t)MTOK * 256 * 2);          // LN out bf16 / conv2 out
  bf16_t* qkvb   = (bf16_t*)alloc((size_t)MTOK * 1536 * 2);         // qkv bf16 / conv2 im2col
  bf16_t* vtb    = (bf16_t*)alloc((size_t)NB * NHEADS * NDH * NL * 2);
  float*  scores = (float*) alloc((size_t)NHEADS * NL * NL * 4);    // per-b / conv3 im2col
  bf16_t* probs  = (bf16_t*)alloc((size_t)NHEADS * NL * NL * 2);    // per-b
  bf16_t* ob     = (bf16_t*)alloc((size_t)MTOK * NINNER * 2);
  bf16_t* ffb    = (bf16_t*)alloc((size_t)MTOK * NMLP * 2);
  float*  y1     = (float*) alloc((size_t)NB * 2 * NL * 256 * 4);
  bf16_t* qkvT   = (bf16_t*)alloc((size_t)NDEPTH * 1536 * 256 * 2);
  bf16_t* outT   = (bf16_t*)alloc((size_t)NDEPTH * 256 * 512 * 2);
  bf16_t* ff1T   = (bf16_t*)alloc((size_t)NDEPTH * 512 * 256 * 2);
  bf16_t* ff2T   = (bf16_t*)alloc((size_t)NDEPTH * 256 * 512 * 2);
  bf16_t* w2b    = (bf16_t*)alloc((size_t)256 * 1536 * 2);
  bf16_t* w3b    = (bf16_t*)alloc((size_t)256 * 768 * 2);
  float*  wmean  = (float*) alloc((size_t)MTOK * 4);
  float*  sgate  = (float*) alloc((size_t)MTOK * 4);
  bf16_t* A2 = qkvb;               // alias: conv2 im2col (dead before qkv use)
  bf16_t* A3 = (bf16_t*)scores;    // alias: conv3 im2col (dead before attention)
  bf16_t* y2b = hb;                // alias: conv2 output (dead before LN1)

  auto blocks = [](long long n) { return (unsigned)((n + 255) / 256); };

  // ---- weight conversion (bf16, Bt = [N,K]) ----
  cvt_bf16_kernel<<<blocks(256LL * 1536), 256, 0, stream>>>(conv2_w, w2b, 256LL * 1536);
  cvt_bf16_kernel<<<blocks(256LL * 768), 256, 0, stream>>>(conv3_w, w3b, 256LL * 768);
  for (int i = 0; i < NDEPTH; ++i) {
    cvt_bf16_T_kernel<<<blocks(256LL * 1536), 256, 0, stream>>>(qkv_w + (size_t)i * 256 * 1536,
                                                                qkvT + (size_t)i * 1536 * 256, 256, 1536);
    cvt_bf16_T_kernel<<<blocks(512LL * 256), 256, 0, stream>>>(out_w + (size_t)i * 512 * 256,
                                                               outT + (size_t)i * 256 * 512, 512, 256);
    cvt_bf16_T_kernel<<<blocks(256LL * 512), 256, 0, stream>>>(ff_w1 + (size_t)i * 256 * 512,
                                                               ff1T + (size_t)i * 512 * 256, 256, 512);
    cvt_bf16_T_kernel<<<blocks(512LL * 256), 256, 0, stream>>>(ff_w2 + (size_t)i * 512 * 256,
                                                               ff2T + (size_t)i * 256 * 512, 512, 256);
  }

  // ---- conv patch embedding ----
  conv1_kernel<<<blocks((long long)NB * 2 * NL * 256), 256, 0, stream>>>(img, conv1_w, conv1_b, y1);
  im2col2_kernel<<<blocks((long long)MTOK * 1536), 256, 0, stream>>>(y1, A2);
  launch_gemm(stream, A2, 1536, 0, w2b, 1536, 0, nullptr, y2b, 256, 0,
              conv2_b, nullptr, nullptr, 0, 1.0f, MTOK, 256, 1536, 1, /*relu*/1);
  im2col3_kernel<<<blocks((long long)MTOK * 768), 256, 0, stream>>>(y2b, A3);
  launch_gemm(stream, A3, 768, 0, w3b, 768, 0, x, nullptr, 256, 0,
              conv3_b, nullptr, nullptr, 0, 1.0f, MTOK, 256, 768, 1, /*relu*/1);

  // ---- transformer layers ----
  const long long LL = (long long)NL * NL;
  for (int i = 0; i < NDEPTH; ++i) {
    layernorm_bf16_kernel<<<MTOK, 256, 0, stream>>>(x, ln1_s + i * 256, ln1_b + i * 256, hb);
    // qkv: [4096,256] x [256,1536]
    launch_gemm(stream, hb, 256, 0, qkvT + (size_t)i * 1536 * 256, 256, 0,
                nullptr, qkvb, 1536, 0, nullptr, nullptr, nullptr, 0,
                1.0f, MTOK, 1536, 256, 1, 0);
    {
      dim3 tg(NL / 64, NHEADS, NB);
      build_vt_kernel<<<tg, 256, 0, stream>>>(qkvb, vtb);
    }
    for (int b = 0; b < NB; ++b) {
      const bf16_t* qbase = qkvb + (size_t)b * NL * 1536;
      // scores[h] = Q*K^T * 1/8 + pos[i,h]   (A = Q rows, Bt = K rows)
      launch_gemm(stream, qbase, 1536, 64, qbase + 512, 1536, 64,
                  scores, nullptr, NL, LL, nullptr, nullptr,
                  pos + (size_t)i * NHEADS * LL, LL,
                  0.125f, NL, NL, NDH, NHEADS, 0);
      softmax_rows_kernel<<<NHEADS * NL, 256, 0, stream>>>(scores, probs);
      // o[h] = P * V   (Bt = V^T [64,1024])
      launch_gemm(stream, probs, NL, LL, vtb + (size_t)b * NHEADS * NDH * NL, NL,
                  (long long)NDH * NL,
                  nullptr, ob + (size_t)b * NL * NINNER, NINNER, 64,
                  nullptr, nullptr, nullptr, 0, 1.0f, NL, NDH, NL, NHEADS, 0);
    }
    // out projection + residual -> x
    launch_gemm(stream, ob, 512, 0, outT + (size_t)i * 256 * 512, 512, 0,
                x, nullptr, 256, 0, out_b + i * 256, x, nullptr, 0,
                1.0f, MTOK, 256, 512, 1, 0);
    // FFN
    layernorm_bf16_kernel<<<MTOK, 256, 0, stream>>>(x, ln2_s + i * 256, ln2_b + i * 256, hb);
    launch_gemm(stream, hb, 256, 0, ff1T + (size_t)i * 512 * 256, 256, 0,
                nullptr, ffb, 512, 0, ff_b1 + i * 512, nullptr, nullptr, 0,
                1.0f, MTOK, 512, 256, 1, /*gelu*/2);
    launch_gemm(stream, ffb, 512, 0, ff2T + (size_t)i * 256 * 512, 512, 0,
                x, nullptr, 256, 0, ff_b2 + i * 256, x, nullptr, 0,
                1.0f, MTOK, 256, 512, 1, 0);
  }

  // ---- SE attention pool ----
  rowmean_kernel<<<MTOK, 256, 0, stream>>>(x, wmean);
  se_kernel<<<NB, 128, 0, stream>>>(wmean, se_w1, se_b1, se_w2, se_b2, sgate);
  pool_out_kernel<<<NB, 256, 0, stream>>>(x, sgate, outp);
}
